// Megnet_EdgeModel_53970559042216
// MI455X (gfx1250) — matrix operand order repaired and checked by reading, not compiled, not run on hardware
//
#include <hip/hip_runtime.h>
#include <hip/hip_bf16.h>

// MegNet edge model on MI455X (gfx1250, wave32, WMMA).
// out = relu(concat(src,dest,edge_attr,u[batch]) @ W1 + b1) @ W2 + b2
// bf16 WMMA (v_wmma_f32_16x16x32_bf16) with f32 accumulation; weight tiles
// pre-swizzled once into WMMA B-fragment layout in d_ws, then staged into LDS
// with GLOBAL_LOAD_ASYNC_TO_LDS_B128 (ASYNCcnt) to avoid VGPR pressure/spills.
// Memory-bound target: ~1.03 GB @ 23.3 TB/s (~44 us).

typedef __attribute__((ext_vector_type(16))) __bf16 v16bf;
typedef __attribute__((ext_vector_type(8)))  float  v8f;
typedef __attribute__((ext_vector_type(4)))  __bf16 bf16x4;
typedef __attribute__((ext_vector_type(4)))  int    v4i;

#define DIM 128
#define TM  128                 // edges per block
#define W1F_ELEMS (512 * 128)   // W1 bf16 fragment store (128 KB)
#define W2F_ELEMS (128 * 128)   // W2 bf16 fragment store (32 KB)

#if defined(__has_builtin)
#if __has_builtin(__builtin_amdgcn_global_load_async_to_lds_b128)
#define HAVE_ASYNC_LDS 1
#endif
#endif

#ifndef HAVE_ASYNC_LDS
#define HAVE_ASYNC_LDS 0
#endif

#if HAVE_ASYNC_LDS
typedef __attribute__((address_space(1))) v4i* gv4p;  // global 128-bit chunk
typedef __attribute__((address_space(3))) v4i* lv4p;  // LDS 128-bit chunk
__device__ __forceinline__ void asyncCopy16(const void* g, void* l) {
  __builtin_amdgcn_global_load_async_to_lds_b128((gv4p)g, (lv4p)l, 0, 0);
}
__device__ __forceinline__ void waitAsync() {
#if __has_builtin(__builtin_amdgcn_s_wait_asynccnt)
  __builtin_amdgcn_s_wait_asynccnt(0);
#else
  asm volatile("s_wait_asynccnt 0" ::: "memory");
#endif
}
#else
__device__ __forceinline__ void waitAsync() {}
#endif

// ---------------------------------------------------------------------------
// One-time weight conversion: f32 -> bf16, scattered into the wave32 WMMA
// B-matrix fragment layout (32x16 per tile):
//   lane = ((k%32 / 8) % 2)*16 + (n % 16); idx = (k%8) + 8*(k%32 / 16).
// Region layout: [kstep][ntile][lane(32)][idx(16)]
// ---------------------------------------------------------------------------
__global__ void convert_weights(const float* __restrict__ W1,
                                const float* __restrict__ W2,
                                __bf16* __restrict__ w1f,
                                __bf16* __restrict__ w2f) {
  int tid = blockIdx.x * blockDim.x + threadIdx.x;
  if (tid < W1F_ELEMS) {
    int k = tid >> 7, n = tid & 127;
    int kstep = k >> 5, kk = k & 31;
    int ntile = n >> 4;
    int lane = (((kk >> 3) & 1) << 4) | (n & 15);
    int idx = (kk & 7) | ((kk >> 4) << 3);
    w1f[((kstep * 8 + ntile) * 32 + lane) * 16 + idx] = (__bf16)W1[tid];
  } else if (tid < W1F_ELEMS + W2F_ELEMS) {
    int t = tid - W1F_ELEMS;
    int k = t >> 7, n = t & 127;
    int kstep = k >> 5, kk = k & 31;
    int ntile = n >> 4;
    int lane = (((kk >> 3) & 1) << 4) | (n & 15);
    int idx = (kk & 7) | ((kk >> 4) << 3);
    w2f[((kstep * 8 + ntile) * 32 + lane) * 16 + idx] = (__bf16)W2[t];
  }
}

// ---------------------------------------------------------------------------
// Main fused kernel. Block = 128 edges x 128 outputs, 256 threads = 8 waves.
// Wave (mi,ni): mi in 0..3 -> rows [mi*32, mi*32+32); ni in 0..1 -> cols
// [ni*64, ni*64+64). 2x4 = 8 f32 16x16 accumulators per wave.
// ---------------------------------------------------------------------------
__global__ __launch_bounds__(256)
void megnet_edge_kernel(const float* __restrict__ src,
                        const float* __restrict__ dst,
                        const float* __restrict__ ea,
                        const float* __restrict__ u,
                        const int*   __restrict__ batch,
                        const __bf16* __restrict__ w1f,
                        const float* __restrict__ b1,
                        const __bf16* __restrict__ w2f,
                        const float* __restrict__ b2,
                        float* __restrict__ out) {
  extern __shared__ char smem[];
  __bf16* aBuf  = (__bf16*)(smem);            // 2 x 4096 bf16 (A frags, dbl buf)
  __bf16* bBuf  = (__bf16*)(smem + 16384);    // 2 x 4096 bf16 (W1 slice frags)
  __bf16* hFrag = (__bf16*)(smem + 32768);    // 16384 bf16 (H in GEMM2 A layout)
  __bf16* w2s   = (__bf16*)(smem + 65536);    // 16384 bf16 (W2 frags)

  const int tid  = threadIdx.x;
  const int lane = tid & 31;
  const int mi   = (tid >> 5) & 3;   // 2 mtiles (32 rows)
  const int ni   = (tid >> 5) >> 2;  // 4 ntiles (64 cols)
  const long long e0 = (long long)blockIdx.x * TM;

  // Stage full pre-swizzled W2 (32 KB) into LDS once (async: no VGPRs used).
#if HAVE_ASYNC_LDS
  {
    const char* g = (const char*)w2f + tid * 16;
    char* l = (char*)w2s + tid * 16;
#pragma unroll
    for (int i = 0; i < 8; ++i)
      asyncCopy16(g + i * 4096, l + i * 4096);
  }
#else
  {
    const float4* s = (const float4*)w2f;
    float4* d = (float4*)w2s;
#pragma unroll
    for (int i = 0; i < (W2F_ELEMS * 2 / 16) / 256; ++i)
      d[tid + i * 256] = s[tid + i * 256];
  }
#endif

  // Accumulators init to b1 (C layout: lane holds one column n = l%16).
  v8f acc[2][4];
#pragma unroll
  for (int nt = 0; nt < 4; ++nt) {
    float bv = b1[ni * 64 + nt * 16 + (lane & 15)];
#pragma unroll
    for (int mt = 0; mt < 2; ++mt)
#pragma unroll
      for (int i = 0; i < 8; ++i) acc[mt][nt][i] = bv;
  }

  const float* srcs[3] = {src, dst, ea};

  // Per-thread constant staging indices for the A fragment scatter.
  // Thread t loads float4 at (row = g*32 + t/8, kloc = (t%8)*4) per k-step.
  // 16-bit A frag layout: lane = ((kloc%16)/8)*16 + r%16,
  //                       idx  = (kloc%8) + 8*(kloc/16)   (4 consecutive).
  const int kloc0 = (tid & 7) << 2;
  const int halfA = (kloc0 >> 3) & 1;
  const int idx0  = (kloc0 & 7) | ((kloc0 >> 4) << 3);
  const int rloc  = tid >> 3;

  float4 ar[4];
#if !HAVE_ASYNC_LDS
  float4 br[2];
#endif

  // Issue W1-slice copy for k-step kk into LDS buffer `buf`.
  auto stageB = [&](int kk, int buf) {
#if HAVE_ASYNC_LDS
    const char* g = (const char*)w1f + (size_t)kk * 8192 + tid * 16;
    char* l = (char*)(bBuf + buf * 4096) + tid * 16;
    asyncCopy16(g, l);
    asyncCopy16(g + 4096, l + 4096);
#else
    (void)buf;
    const float4* s = (const float4*)(w1f + (size_t)kk * 4096);
    br[0] = s[tid];
    br[1] = s[tid + 256];
#endif
  };

  auto loadStage = [&](int kk) {
    const int sidx = kk >> 2;                      // which concat block
    const int colbase = ((kk & 3) << 5) + kloc0;   // column in that block
#pragma unroll
    for (int g = 0; g < 4; ++g) {
      long long r = e0 + g * 32 + rloc;
      const float* rowp = (sidx < 3) ? (srcs[sidx] + (size_t)r * DIM)
                                     : (u + (size_t)batch[r] * DIM);
      ar[g] = *(const float4*)(rowp + colbase);
    }
  };

  auto storeStage = [&](int buf) {
    __bf16* ab = aBuf + buf * 4096;
#pragma unroll
    for (int g = 0; g < 4; ++g) {
      int rr = g * 32 + rloc;
      bf16x4 p;
      p.x = (__bf16)ar[g].x; p.y = (__bf16)ar[g].y;
      p.z = (__bf16)ar[g].z; p.w = (__bf16)ar[g].w;
      *(bf16x4*)(ab + (((rr >> 4) * 32 + halfA * 16 + (rr & 15)) * 16 + idx0)) = p;
    }
#if !HAVE_ASYNC_LDS
    float4* d = (float4*)(bBuf + buf * 4096);
    d[tid] = br[0];
    d[tid + 256] = br[1];
#endif
  };

  auto computeStep = [&](int buf) {
    const __bf16* ab = aBuf + buf * 4096;
    const __bf16* bb = bBuf + buf * 4096;
    v16bf aF[2], bF[4];
#pragma unroll
    for (int mt = 0; mt < 2; ++mt)
      aF[mt] = *(const v16bf*)(ab + ((mi * 2 + mt) * 32 + lane) * 16);
#pragma unroll
    for (int nt = 0; nt < 4; ++nt)
      bF[nt] = *(const v16bf*)(bb + ((ni * 4 + nt) * 32 + lane) * 16);
#pragma unroll
    for (int mt = 0; mt < 2; ++mt)
#pragma unroll
      for (int nt = 0; nt < 4; ++nt)
        acc[mt][nt] = __builtin_amdgcn_wmma_f32_16x16x32_bf16(
            false, aF[mt], false, bF[nt], (short)0, acc[mt][nt], false, false);
  };

  // ---------------- GEMM1: K = 512, 16 k-steps, double buffered ------------
  loadStage(0);
  stageB(0, 0);
  storeStage(0);
  waitAsync();
  __syncthreads();
  for (int kk = 0; kk < 16; ++kk) {
    const int cur = kk & 1;
    if (kk < 15) {
      loadStage(kk + 1);       // global loads overlap compute
      stageB(kk + 1, cur ^ 1); // async weight slice straight into LDS
    }
    computeStep(cur);
    if (kk < 15) storeStage(cur ^ 1);
    waitAsync();
    __syncthreads();
  }

  // ---------------- ReLU + bf16 H into GEMM2 A-fragment layout -------------
#pragma unroll
  for (int mt = 0; mt < 2; ++mt) {
    const int mtile2 = mi * 2 + mt;
    const int rb = (lane >> 4) << 3;     // 8*(l/16): row sub-offset
#pragma unroll
    for (int nt = 0; nt < 4; ++nt) {
      const int c = (ni * 4 + nt) * 16 + (lane & 15);  // K index for GEMM2
      const int ks2 = c >> 5, kk2 = c & 31;
      const int half = (kk2 >> 3) & 1;
      const int idx = (kk2 & 7) | ((kk2 >> 4) << 3);
      __bf16* hp = hFrag + ((ks2 * 8 + mtile2) * 32 + half * 16) * 16 + idx;
#pragma unroll
      for (int vg = 0; vg < 8; ++vg) {
        float v = acc[mt][nt][vg];
        v = v > 0.f ? v : 0.f;
        hp[(rb + vg) * 16] = (__bf16)v;
      }
    }
  }
  __syncthreads();

  // ---------------- GEMM2: K = 128, 4 k-steps from LDS ---------------------
  v8f acc2[2][4];
#pragma unroll
  for (int nt = 0; nt < 4; ++nt) {
    float bv = b2[ni * 64 + nt * 16 + (lane & 15)];
#pragma unroll
    for (int mt = 0; mt < 2; ++mt)
#pragma unroll
      for (int i = 0; i < 8; ++i) acc2[mt][nt][i] = bv;
  }
#pragma unroll
  for (int ks = 0; ks < 4; ++ks) {
    v16bf aF[2], bF[4];
#pragma unroll
    for (int mt = 0; mt < 2; ++mt)
      aF[mt] = *(const v16bf*)(hFrag + ((ks * 8 + mi * 2 + mt) * 32 + lane) * 16);
#pragma unroll
    for (int nt = 0; nt < 4; ++nt)
      bF[nt] = *(const v16bf*)(w2s + ((ks * 8 + ni * 4 + nt) * 32 + lane) * 16);
#pragma unroll
    for (int mt = 0; mt < 2; ++mt)
#pragma unroll
      for (int nt = 0; nt < 4; ++nt)
        acc2[mt][nt] = __builtin_amdgcn_wmma_f32_16x16x32_bf16(
            false, aF[mt], false, bF[nt], (short)0, acc2[mt][nt], false, false);
  }

  // ---------------- Store output (f32, coalesced per 16-lane half) ---------
#pragma unroll
  for (int mt = 0; mt < 2; ++mt) {
    long long rbase = e0 + (mi * 2 + mt) * 16 + ((lane >> 4) << 3);
#pragma unroll
    for (int vg = 0; vg < 8; ++vg) {
      float* op = out + (size_t)(rbase + vg) * DIM + (lane & 15);
#pragma unroll
      for (int nt = 0; nt < 4; ++nt)
        op[(ni * 4 + nt) * 16] = acc2[mt][nt][vg];
    }
  }
}

// ---------------------------------------------------------------------------
extern "C" void kernel_launch(void* const* d_in, const int* in_sizes, int n_in,
                              void* d_out, int out_size, void* d_ws, size_t ws_size,
                              hipStream_t stream) {
  const float* src  = (const float*)d_in[0];
  const float* dst  = (const float*)d_in[1];
  const float* ea   = (const float*)d_in[2];
  const float* u    = (const float*)d_in[3];
  const int*   bat  = (const int*)d_in[4];
  const float* W1   = (const float*)d_in[5];
  const float* b1   = (const float*)d_in[6];
  const float* W2   = (const float*)d_in[7];
  const float* b2   = (const float*)d_in[8];
  float* out = (float*)d_out;

  __bf16* w1f = (__bf16*)d_ws;
  __bf16* w2f = w1f + W1F_ELEMS;

  const int E = in_sizes[0] / DIM;  // 400000

  // One-time weight convert+swizzle into d_ws (bf16 fragment layout).
  const int conv_threads = W1F_ELEMS + W2F_ELEMS;
  convert_weights<<<(conv_threads + 255) / 256, 256, 0, stream>>>(W1, W2, w1f, w2f);

  // Fused edge-model GEMM pipeline: 128 edges per block.
  const int grid = E / TM;  // 3125
  const size_t lds_bytes = 98304;  // 96 KB -> 3 workgroups per 320 KB WGP
  megnet_edge_kernel<<<grid, 256, lds_bytes, stream>>>(
      src, dst, ea, u, bat, w1f, b1, w2f, b2, out);
}
